// GroupedAttentionLayer_20667382629171
// MI455X (gfx1250) — matrix operand order repaired and checked by reading, not compiled, not run on hardware
//
#include <hip/hip_runtime.h>
#include <hip/hip_bf16.h>

// ---------------- problem constants ----------------
#define HDIM 4096
#define NH   32
#define NKV  8
#define DH   128
#define GQ   (NH / NKV)          // 4
#define SEQ  2048
#define BATCH 2
#define NTOK (BATCH * SEQ)       // 4096
#define QKVN (NH*DH + 2*NKV*DH)  // 6144
#define S16  (SEQ / 16)          // 128

typedef __attribute__((ext_vector_type(16))) __bf16 v16bf;
typedef __attribute__((ext_vector_type(8)))  __bf16 v8bf;
typedef __attribute__((ext_vector_type(8)))  float  v8f;

// ---------------- WMMA helpers ----------------
__device__ __forceinline__ v8f wmma_bf16(v16bf a, v16bf b, v8f c) {
  // D(16x16,f32) = A(16x32,bf16) * B(32x16,bf16) + C
  return __builtin_amdgcn_wmma_f32_16x16x32_bf16(
      /*neg_a=*/false, a, /*neg_b=*/false, b,
      /*c_mod=*/(short)0, c, /*reuse_a=*/false, /*reuse_b=*/false);
}

__device__ __forceinline__ v8f vzero8() {
  v8f z = {0.f,0.f,0.f,0.f,0.f,0.f,0.f,0.f};
  return z;
}

// A-tile (16x32 bf16, row-major source [., ld]):
// lane 0-15: M=lane, elems 0..7 -> K=0..7, elems 8..15 -> K=16..23
// lane 16-31: M=lane-16, elems 0..7 -> K=8..15, elems 8..15 -> K=24..31
__device__ __forceinline__ v16bf load_a_tile(const __bf16* base, int ld,
                                             int m0, int kc, int lane) {
  int m  = m0 + (lane & 15);
  int hi = lane >> 4;
  const __bf16* p = base + (size_t)m * ld + kc + hi * 8;
  union { v16bf v; v8bf h[2]; } u;
  u.h[0] = *reinterpret_cast<const v8bf*>(p);        // K = kc+hi*8 .. +7
  u.h[1] = *reinterpret_cast<const v8bf*>(p + 16);   // K = kc+16+hi*8 .. +7
  return u.v;
}

// B-tile (32x16 bf16) from a [N, K]-major array (i.e. B^T row-major):
// lane 0-15: N=lane, elems = K 0..7,16..23 ; lane 16-31: N=lane-16, elems = K 8..15,24..31
__device__ __forceinline__ v16bf load_b_tile(const __bf16* base, int ld,
                                             int n0, int kc, int lane) {
  int n  = n0 + (lane & 15);
  int kb = (lane >> 4) * 8;
  const __bf16* p = base + (size_t)n * ld + kc + kb;
  union { v16bf v; v8bf h[2]; } u;
  u.h[0] = *reinterpret_cast<const v8bf*>(p);
  u.h[1] = *reinterpret_cast<const v8bf*>(p + 16);
  return u.v;
}

// Same B-tile striping but sourced from LDS (base = smem + buffer offset).
__device__ __forceinline__ v16bf lds_b_tile(const char* base, int rowBytes,
                                            int n0, int kc, int lane) {
  int n  = n0 + (lane & 15);
  int kb = (lane >> 4) * 8;
  const char* p = base + n * rowBytes + (kc + kb) * 2;
  union { v16bf v; v8bf h[2]; } u;
  u.h[0] = *reinterpret_cast<const v8bf*>(p);
  u.h[1] = *reinterpret_cast<const v8bf*>(p + 32);
  return u.v;
}

// ---------------- CDNA5 async global->LDS copy ----------------
// global_load_async_to_lds_b128: VDST = per-lane LDS byte offset,
// VADDR = per-lane 64-bit global address. Tracked by ASYNCcnt.
__device__ __forceinline__ void async_ld_b128(unsigned lds_off, const char* g) {
  asm volatile("global_load_async_to_lds_b128 %0, %1, off"
               :: "v"(lds_off), "v"(g) : "memory");
}
// Stage 32 rows x 256B (one 32-key K panel, contiguous rows): 16 asyncs/wave.
__device__ __forceinline__ void stage_k(const char* kbytes, unsigned buf, int lane) {
#pragma unroll
  for (int t = 0; t < 16; ++t) {
    unsigned off = (unsigned)(lane + 32 * t) * 16;   // 8192 B total
    async_ld_b128(buf + off, kbytes + off);
  }
}
// Stage 128 rows x 64B (one Vt panel: d-rows, 32 key cols): 16 asyncs/wave.
__device__ __forceinline__ void stage_v(const char* vbytes, unsigned buf, int lane) {
#pragma unroll
  for (int t = 0; t < 16; ++t) {
    int c = lane + 32 * t;                 // 0..511 chunks of 16B
    int row = c >> 2, within = (c & 3) * 16;
    async_ld_b128(buf + (unsigned)(row * 64 + within),
                  vbytes + (size_t)row * (SEQ * 2) + within);
  }
}

// ---------------- conversion kernels ----------------
__global__ void cvt_bf16(const float* __restrict__ in,
                         __bf16* __restrict__ out, long n) {
  long i = blockIdx.x * (long)blockDim.x + threadIdx.x;
  long stride = gridDim.x * (long)blockDim.x;
  for (; i < n; i += stride) out[i] = (__bf16)in[i];
}

// in: [K][N] fp32 row-major  ->  out: [N][K] bf16 row-major
__global__ void transpose_cvt_bf16(const float* __restrict__ in,
                                   __bf16* __restrict__ out, int K, int N) {
  long i = blockIdx.x * (long)blockDim.x + threadIdx.x;
  long total = (long)K * N;
  long stride = gridDim.x * (long)blockDim.x;
  for (; i < total; i += stride) {
    long k = i / N, n = i % N;
    out[n * (long)K + k] = (__bf16)in[i];
  }
}

// ---------------- WMMA GEMM: C[M,N] = A[M,K] * Bt[N,K]^T ----------------
// 8 waves/block (2 rows x 4 cols); each wave computes 32x64 via 2x4 WMMA
// tiles -> 8 WMMAs per 12 b128 loads per K-step. Block tile = 64x256.
__global__ __launch_bounds__(256) void gemm_bf16_wmma(
    const __bf16* __restrict__ A, const __bf16* __restrict__ Bt,
    float* __restrict__ C, int M, int N, int K) {
  const int lane = threadIdx.x & 31;
  const int wave = threadIdx.x >> 5;                    // 0..7
  const int m0 = blockIdx.y * 64 + (wave >> 2) * 32;    // 2 wave-rows
  const int n0 = blockIdx.x * 256 + (wave & 3) * 64;    // 4 wave-cols

  const __bf16* Ar = A  + (size_t)(m0 + (lane & 15)) * K;
  const __bf16* Br = Bt + (size_t)(n0 + (lane & 15)) * K;

  v8f acc[2][4];
#pragma unroll
  for (int i = 0; i < 2; ++i)
#pragma unroll
    for (int j = 0; j < 4; ++j) acc[i][j] = vzero8();

  for (int kc = 0; kc < K; kc += 32) {
    // prefetch ahead (global_prefetch_b8, speculative)
    __builtin_prefetch(Ar + kc + 256, 0, 1);
    __builtin_prefetch(Br + kc + 256, 0, 1);

    v16bf a0 = load_a_tile(A, K, m0,      kc, lane);
    v16bf a1 = load_a_tile(A, K, m0 + 16, kc, lane);
    v16bf b[4];
#pragma unroll
    for (int j = 0; j < 4; ++j) b[j] = load_b_tile(Bt, K, n0 + 16 * j, kc, lane);
#pragma unroll
    for (int j = 0; j < 4; ++j) {
      acc[0][j] = wmma_bf16(a0, b[j], acc[0][j]);
      acc[1][j] = wmma_bf16(a1, b[j], acc[1][j]);
    }
  }

  const int cn = lane & 15, hi = lane >> 4;
#pragma unroll
  for (int i = 0; i < 2; ++i) {
#pragma unroll
    for (int j = 0; j < 4; ++j) {
#pragma unroll
      for (int r = 0; r < 8; ++r) {
        int row = m0 + 16 * i + r + 8 * hi;
        C[(size_t)row * N + n0 + 16 * j + cn] = acc[i][j][r];
      }
    }
  }
}

// ---------------- RMSNorm + RoPE + pack (Q or K) ----------------
// block = 128 threads = one (token, head) row of D=128.
// out layout: [B, nheads, S, D] bf16 (head-major panels for attention).
__global__ __launch_bounds__(128) void qknorm_rope_pack(
    const float* __restrict__ qkv, int colOff, int nheads,
    const float* __restrict__ w, const int* __restrict__ pos,
    __bf16* __restrict__ out) {
  const int d = threadIdx.x;
  int idx = blockIdx.x;                     // b*S*nheads + s*nheads + h
  const int h = idx % nheads;
  const int t = idx / nheads;               // b*S + s
  const int s = t % SEQ, b = t / SEQ;

  __shared__ float red[128];
  __shared__ float nv[128];

  float x = qkv[(size_t)t * QKVN + colOff + h * DH + d];
  red[d] = x * x;
  __syncthreads();
  for (int o = 64; o > 0; o >>= 1) {
    if (d < o) red[d] += red[d + o];
    __syncthreads();
  }
  float rs = rsqrtf(red[0] * (1.0f / DH) + 1e-6f);
  float xn = x * rs * w[d];
  nv[d] = xn;
  __syncthreads();
  float rot = (d < 64) ? -nv[d + 64] : nv[d - 64];
  float p = (float)pos[t];
  float invf = __powf(10000.0f, -(float)(d & 63) * (1.0f / 64.0f));
  float sn, cs;
  __sincosf(p * invf, &sn, &cs);
  float y = xn * cs + rot * sn;
  out[(((size_t)b * nheads + h) * SEQ + s) * DH + d] = (__bf16)y;
}

// ---------------- V pack: [B,S,KV,D] fp32 -> [B,KV,D,S] bf16 ----------------
__global__ void v_pack(const float* __restrict__ qkv, __bf16* __restrict__ vt) {
  long i = blockIdx.x * (long)blockDim.x + threadIdx.x;
  if (i >= (long)NTOK * NKV * DH) return;
  int d  = i % DH;
  int r  = i / DH;
  int kv = r % NKV;
  int t  = r / NKV;
  int s = t % SEQ, b = t / SEQ;
  float x = qkv[(size_t)t * QKVN + (NH + NKV) * DH + kv * DH + d];
  vt[(((size_t)b * NKV + kv) * DH + d) * SEQ + s] = (__bf16)x;
}

// ---------------- single-wave flash attention ----------------
// One wave owns 16 query rows of one head; 32 keys per step. K and Vt panels
// are double-buffered in LDS via async global->LDS copies (ASYNCcnt).
// LDS map (dynamic): [0,8K)=Kbuf0 [8K,16K)=Kbuf1 [16K,24K)=Vbuf0
//                    [24K,32K)=Vbuf1 [32K,34K)=P scratch
__global__ __launch_bounds__(32) void flash_attn(
    const __bf16* __restrict__ Q, const __bf16* __restrict__ Kc,
    const __bf16* __restrict__ Vt, __bf16* __restrict__ O) {
  extern __shared__ char smem[];
  float* Pls = (float*)(smem + 32768);

  const int lane = threadIdx.x;
  int id = blockIdx.x;
  const int qt = id % S16; id /= S16;
  const int h  = id % NH;
  const int b  = id / NH;
  const int kv = h / GQ;
  const int m0 = qt * 16;

  const __bf16* Qb = Q  + (((size_t)b * NH  + h ) * SEQ + m0) * DH;
  const char* Kbytes = (const char*)(Kc + ((size_t)b * NKV + kv) * SEQ * DH);
  const char* Vbytes = (const char*)(Vt + ((size_t)b * NKV + kv) * DH * SEQ);

  // Q tile held resident: 16x128 = 4 A-tiles of 16x32
  v16bf qa[4];
#pragma unroll
  for (int c = 0; c < 4; ++c) qa[c] = load_a_tile(Qb, DH, 0, c * 32, lane);

  v8f occ[8];
#pragma unroll
  for (int c = 0; c < 8; ++c) occ[c] = vzero8();
  float mprev[8], lsum[8];
#pragma unroll
  for (int j = 0; j < 8; ++j) { mprev[j] = -1e30f; lsum[j] = 0.f; }

  const int hi = lane >> 4, ln = lane & 15;
  const float scale = 0.08838834764831845f;   // 1/sqrt(128)
  const int nblk = (m0 + 16 + 31) / 32;       // causal upper bound

  // preload block 0 into buffer 0 (32 async instructions per batch)
  stage_k(Kbytes, 0u, lane);
  stage_v(Vbytes, 16384u, lane);

  for (int blk = 0; blk < nblk; ++blk) {
    const int j0 = blk * 32;
    const unsigned kbuf = (blk & 1) ? 8192u : 0u;
    const unsigned vbuf = (blk & 1) ? 24576u : 16384u;

    if (blk + 1 < nblk) {
      // issue next batch into the other buffer, then wait only for the
      // previous batch (<=32 outstanding; async loads retire in order)
      stage_k(Kbytes + (size_t)(j0 + 32) * 256, kbuf ^ 8192u, lane);
      stage_v(Vbytes + (size_t)(j0 + 32) * 2,  vbuf ^ 8192u, lane);
      asm volatile("s_wait_asynccnt 0x20" ::: "memory");
    } else {
      asm volatile("s_wait_asynccnt 0x0" ::: "memory");
    }

    // S = Q * K^T : two 16x16 score tiles, K-dim 128 = 4 chunks (from LDS)
    v8f s0 = vzero8(), s1 = vzero8();
#pragma unroll
    for (int c = 0; c < 4; ++c) {
      v16bf k0 = lds_b_tile(smem + kbuf, 256, 0,  c * 32, lane);
      v16bf k1 = lds_b_tile(smem + kbuf, 256, 16, c * 32, lane);
      s0 = wmma_bf16(qa[c], k0, s0);
      s1 = wmma_bf16(qa[c], k1, s1);
    }

    // scale + causal mask + online softmax. C-layout: row = m0+j+8*hi,
    // col = j0 + ln (+16 for s1). Row reductions stay inside the 16-lane
    // half via xor shuffles 1,2,4,8 (wave32).
    float p0[8], p1[8], corr[8];
#pragma unroll
    for (int j = 0; j < 8; ++j) {
      int row = m0 + j + 8 * hi;
      int c0 = j0 + ln, c1 = c0 + 16;
      float v0 = (c0 <= row) ? s0[j] * scale : -1e30f;
      float v1 = (c1 <= row) ? s1[j] * scale : -1e30f;
      float mx = fmaxf(v0, v1);
#pragma unroll
      for (int o = 1; o < 16; o <<= 1) mx = fmaxf(mx, __shfl_xor(mx, o, 32));
      float mnew = fmaxf(mprev[j], mx);
      corr[j] = __expf(mprev[j] - mnew);
      mprev[j] = mnew;
      float e0 = __expf(v0 - mnew), e1 = __expf(v1 - mnew);
      float ps = e0 + e1;
#pragma unroll
      for (int o = 1; o < 16; o <<= 1) ps += __shfl_xor(ps, o, 32);
      lsum[j] = lsum[j] * corr[j] + ps;
      p0[j] = e0; p1[j] = e1;
    }
#pragma unroll
    for (int c = 0; c < 8; ++c)
#pragma unroll
      for (int j = 0; j < 8; ++j) occ[c][j] *= corr[j];

    // Re-stripe P from C-layout to A-layout through LDS (single wave).
    __syncthreads();
#pragma unroll
    for (int j = 0; j < 8; ++j) {
      Pls[(j + 8 * hi) * 32 + ln]      = p0[j];
      Pls[(j + 8 * hi) * 32 + 16 + ln] = p1[j];
    }
    __syncthreads();
    v16bf pa;
#pragma unroll
    for (int e = 0; e < 16; ++e) {
      int k = (e < 8) ? (hi * 8 + e) : (16 + hi * 8 + (e - 8));
      pa[e] = (__bf16)Pls[ln * 32 + k];
    }

    // O += P(16x32) * V(32x128): 8 d-chunks; Vt panel in LDS, k-contiguous
#pragma unroll
    for (int c = 0; c < 8; ++c) {
      v16bf bv = lds_b_tile(smem + vbuf, 64, c * 16, 0, lane);
      occ[c] = wmma_bf16(pa, bv, occ[c]);
    }
  }

  // normalize and store attn output in [B, S, H*D] bf16
#pragma unroll
  for (int c = 0; c < 8; ++c) {
#pragma unroll
    for (int j = 0; j < 8; ++j) {
      int row = m0 + j + 8 * hi;
      float y = occ[c][j] / lsum[j];
      O[((size_t)(b * SEQ + row)) * (NH * DH) + h * DH + c * 16 + ln] = (__bf16)y;
    }
  }
}

// ---------------- host launch ----------------
extern "C" void kernel_launch(void* const* d_in, const int* in_sizes, int n_in,
                              void* d_out, int out_size, void* d_ws, size_t ws_size,
                              hipStream_t stream) {
  const float* x   = (const float*)d_in[0];
  const float* wq  = (const float*)d_in[1];
  const float* wk  = (const float*)d_in[2];
  const float* wv  = (const float*)d_in[3];
  const float* wo  = (const float*)d_in[4];
  const float* qw  = (const float*)d_in[5];
  const float* kw  = (const float*)d_in[6];
  const int*   pos = (const int*)d_in[7];
  float* out = (float*)d_out;
  char* ws = (char*)d_ws;

  // workspace layout (bytes)
  __bf16* Xbf  = (__bf16*)(ws + 0);                    // 32 MB  [NTOK, HID]
  __bf16* Wqkv = (__bf16*)(ws + (size_t)33554432);     // 48 MB  [6144, HID] (B^T)
  __bf16* Wot  = (__bf16*)(ws + (size_t)83886080);     // 32 MB  [HID, H*D]  (B^T)
  float*  QKV  = (float*)(ws + (size_t)117440512);     // 96 MB  [NTOK, 6144]
  __bf16* Qbf  = (__bf16*)(ws + (size_t)218103808);    // 32 MB  [B,H,S,D]
  __bf16* Kbf  = (__bf16*)(ws + (size_t)251658240);    //  8 MB  [B,KV,S,D]
  __bf16* Vtb  = (__bf16*)(ws + (size_t)260046848);    //  8 MB  [B,KV,D,S]
  __bf16* Attn = (__bf16*)(ws + (size_t)268435456);    // 32 MB  [B,S,H*D]

  // 1) bf16 conversion + weight transposes
  cvt_bf16<<<2048, 256, 0, stream>>>(x, Xbf, (long)NTOK * HDIM);
  transpose_cvt_bf16<<<2048, 256, 0, stream>>>(wq, Wqkv, HDIM, NH * DH);
  transpose_cvt_bf16<<<2048, 256, 0, stream>>>(
      wk, Wqkv + (size_t)NH * DH * HDIM, HDIM, NKV * DH);
  transpose_cvt_bf16<<<2048, 256, 0, stream>>>(
      wv, Wqkv + (size_t)(NH + NKV) * DH * HDIM, HDIM, NKV * DH);
  transpose_cvt_bf16<<<2048, 256, 0, stream>>>(wo, Wot, NH * DH, HDIM);

  // 2) fused QKV projection: [4096, 6144] = X * [wq|wk|wv]
  dim3 g1(QKVN / 256, NTOK / 64);
  gemm_bf16_wmma<<<g1, 256, 0, stream>>>(Xbf, Wqkv, QKV, NTOK, QKVN, HDIM);

  // 3) per-head RMSNorm + RoPE + pack
  qknorm_rope_pack<<<NTOK * NH,  128, 0, stream>>>(QKV, 0,       NH,  qw, pos, Qbf);
  qknorm_rope_pack<<<NTOK * NKV, 128, 0, stream>>>(QKV, NH * DH, NKV, kw, pos, Kbf);
  v_pack<<<(NTOK * NKV * DH) / 256, 256, 0, stream>>>(QKV, Vtb);

  // 4) causal GQA flash attention (34816 B dynamic LDS per block)
  flash_attn<<<BATCH * NH * S16, 32, 34816, stream>>>(Qbf, Kbf, Vtb, Attn);

  // 5) output projection -> fp32 d_out
  dim3 g2(HDIM / 256, NTOK / 64);
  gemm_bf16_wmma<<<g2, 256, 0, stream>>>(Attn, Wot, out, NTOK, HDIM, HDIM);
}
